// SparseAttentionAggregator_13065290514527
// MI455X (gfx1250) — matrix operand order repaired and checked by reading, not compiled, not run on hardware
//
#include <hip/hip_runtime.h>
#include <hip/hip_bf16.h>

// ---------------------------------------------------------------------------
// SparseAttentionAggregator on MI455X (gfx1250, wave32, WMMA)
//   Phase 1: Q/K/V = X@W + b   -> bf16 WMMA 16x16x32, 64x64 block tile,
//            2 WMMAs/wave per K-step, float4 staging (b128 global loads)
//   Phase 2: banded attention as two dense WMMA GEMMs over an 80(->96)-slot
//            absolute key range + masked softmax + fused mean-pool
//   Phase 3: relu(pool/S @ Wfc + bfc)  (tiny, VALU)
// ---------------------------------------------------------------------------

#define BATCH 2
#define SEQ   1024
#define DM    768
#define NH    12
#define HD    64
#define WIN   32

typedef __attribute__((ext_vector_type(16))) __bf16          v16bf;
typedef __attribute__((ext_vector_type(16))) unsigned short  v16us;
typedef __attribute__((ext_vector_type(8)))  float           v8f;

__device__ __forceinline__ unsigned int bf16_rne(float f) {
  unsigned int u = __builtin_bit_cast(unsigned int, f);
  return (u + 0x7FFFu + ((u >> 16) & 1u)) >> 16;
}
// pack two floats into one dword of two bf16 (lo, hi)
__device__ __forceinline__ unsigned int pack2(float lo, float hi) {
  return bf16_rne(lo) | (bf16_rne(hi) << 16);
}

__device__ __forceinline__ v8f wmma_bf16(const v16us& a, const v16us& b, v8f c) {
  return __builtin_amdgcn_wmma_f32_16x16x32_bf16(
      false, __builtin_bit_cast(v16bf, a),
      false, __builtin_bit_cast(v16bf, b),
      (short)0, c, false, false);
}

// ---------------------------------------------------------------------------
__global__ void zero_kernel(float* __restrict__ p, int n) {
  int i = blockIdx.x * blockDim.x + threadIdx.x;
  if (i < n) p[i] = 0.0f;
}

// ---------------------------------------------------------------------------
// Kernel 1: QKV projection. Y = X@W + b (M=2048, N=768, K=768), 3x.
// Block = 256 threads (8 waves) computes a 64(M) x 64(N) tile.
// Waves arranged 2(M) x 4(N); each wave owns two 16x16 tiles (shared B frag).
// ---------------------------------------------------------------------------
__global__ __launch_bounds__(256)
void qkv_proj_kernel(const float* __restrict__ X,
                     const float* __restrict__ Wq, const float* __restrict__ bq,
                     const float* __restrict__ Wk, const float* __restrict__ bk,
                     const float* __restrict__ Wv, const float* __restrict__ bv,
                     float* __restrict__ qo, float* __restrict__ ko,
                     float* __restrict__ vo)
{
  const int which = blockIdx.z;
  const float* Wm = (which == 0) ? Wq : (which == 1) ? Wk : Wv;
  const float* bm = (which == 0) ? bq : (which == 1) ? bk : bv;
  float*       Y  = (which == 0) ? qo : (which == 1) ? ko : vo;

  const int m0   = blockIdx.x * 64;
  const int n0   = blockIdx.y * 64;
  const int tid  = threadIdx.x;
  const int lane = tid & 31;
  const int wid  = tid >> 5;      // 0..7
  const int wm   = wid >> 2;      // 0..1  -> M rows [wm*32, wm*32+32)
  const int wn   = wid & 3;       // 0..3  -> N cols [wn*16, wn*16+16)

  // stride 40 halfwords: rows 80B apart (16B aligned), conflict-free frag reads
  __shared__ __align__(16) unsigned short Xs[64 * 40];  // 64(M) x 32(K) bf16
  __shared__ __align__(16) unsigned short Ws[64 * 40];  // 64(N) x 32(K) bf16

  // staging coordinates (fixed per thread)
  const int xr  = tid >> 2;         // X: 64 rows, 4 threads/row
  const int xc8 = (tid & 3) * 8;    //    8 consecutive K each -> 2x float4 load
  const int wnc = tid & 63;         // W: column n (coalesced across lanes)
  const int wc8 = (tid >> 6) * 8;   //    8 K rows each

  v8f acc0 = {}, acc1 = {};

  for (int kk = 0; kk < DM; kk += 32) {
    // ---- stage X chunk (64x32): two b128 loads, one b128 LDS store ----
    const float* xp = X + (size_t)(m0 + xr) * DM + kk + xc8;
    float4 x0 = *(const float4*)xp;
    float4 x1 = *(const float4*)(xp + 4);
    uint4 xo;
    xo.x = pack2(x0.x, x0.y); xo.y = pack2(x0.z, x0.w);
    xo.z = pack2(x1.x, x1.y); xo.w = pack2(x1.z, x1.w);
    *(uint4*)&Xs[xr * 40 + xc8] = xo;

    // ---- stage W chunk (32K x 64N), transposed to [n][k]: 8 coalesced b32
    //      loads batched into registers, one b128 LDS store ----
    const float* wp = Wm + (size_t)(kk + wc8) * DM + n0 + wnc;
    float wv[8];
    #pragma unroll
    for (int j = 0; j < 8; ++j) wv[j] = wp[(size_t)j * DM];
    uint4 wo;
    wo.x = pack2(wv[0], wv[1]); wo.y = pack2(wv[2], wv[3]);
    wo.z = pack2(wv[4], wv[5]); wo.w = pack2(wv[6], wv[7]);
    *(uint4*)&Ws[wnc * 40 + wc8] = wo;

    __syncthreads();

    // ---- fragments: A per 16-bit 16x32 layout, B per 32x16 layout ----
    const int arow0 = wm * 32 + (lane & 15);
    const int akb   = (lane < 16) ? 0 : 8;
    v16us a0, a1, bf;
    #pragma unroll
    for (int j = 0; j < 8; ++j) {
      a0[j]     = Xs[arow0 * 40 + akb + j];
      a0[8 + j] = Xs[arow0 * 40 + akb + 16 + j];
      a1[j]     = Xs[(arow0 + 16) * 40 + akb + j];
      a1[8 + j] = Xs[(arow0 + 16) * 40 + akb + 16 + j];
    }
    const int bcol = 16 * wn + (lane & 15);
    const int bkb  = (lane < 16) ? 0 : 8;
    #pragma unroll
    for (int j = 0; j < 16; ++j)
      bf[j] = Ws[bcol * 40 + bkb + j + (j & 8)];

    acc0 = wmma_bf16(a0, bf, acc0);
    acc1 = wmma_bf16(a1, bf, acc1);

    __syncthreads();
  }

  // C layout: VGPR r, lanes 0-15 -> (M=r, N=lane); lanes 16-31 -> (M=r+8, N=lane-16)
  const int nOut   = n0 + 16 * wn + (lane & 15);
  const int mBase0 = m0 + wm * 32 + ((lane < 16) ? 0 : 8);
  const float bias = bm[nOut];
  #pragma unroll
  for (int r = 0; r < 8; ++r) {
    Y[(size_t)(mBase0 + r) * DM + nOut]      = acc0[r] + bias;
    Y[(size_t)(mBase0 + 16 + r) * DM + nOut] = acc1[r] + bias;
  }
}

// ---------------------------------------------------------------------------
// Kernel 2: banded attention for a 16-query tile + fused mean-pool.
// One wave per (b, h, s-tile). Keys span absolute [s0-32, s0+47] = 80 slots,
// padded to 96 (3 k-steps of 32).
// ---------------------------------------------------------------------------
__global__ __launch_bounds__(32)
void attn_pool_kernel(const float* __restrict__ q, const float* __restrict__ k,
                      const float* __restrict__ v, float* __restrict__ pooled)
{
  const int st = blockIdx.x, h = blockIdx.y, b = blockIdx.z;
  const int s0     = st * 16;
  const int kstart = s0 - WIN;        // absolute position of relative slot 0
  const int lane   = threadIdx.x;

  __shared__ __align__(16) unsigned short Qs[16 * 72];    // 16(M) x 64(d)
  __shared__ __align__(16) unsigned short Ks[96 * 72];    // 96(pos) x 64(d)
  __shared__ __align__(16) unsigned short Vst[64 * 104];  // 64(d) x 96(pos)
  __shared__ __align__(16) unsigned short Ps[16 * 104];   // 16(M) x 96(pos)

  const float* qb = q + (size_t)b * SEQ * DM + h * HD;
  const float* kb = k + (size_t)b * SEQ * DM + h * HD;
  const float* vb = v + (size_t)b * SEQ * DM + h * HD;

  // ---- stage Q (16x64): float4 loads, uint2 LDS stores ----
  #pragma unroll
  for (int it = 0; it < 8; ++it) {
    int lin = it * 128 + lane * 4;
    int r = lin >> 6, d = lin & 63;
    float4 qv = *(const float4*)(qb + (size_t)(s0 + r) * DM + d);
    uint2 o; o.x = pack2(qv.x, qv.y); o.y = pack2(qv.z, qv.w);
    *(uint2*)&Qs[r * 72 + d] = o;
  }
  // ---- stage K and V (96x64): clamped addresses + mask-by-multiply ----
  #pragma unroll 4
  for (int it = 0; it < 48; ++it) {
    int lin = it * 128 + lane * 4;
    int p = lin >> 6, d = lin & 63;
    int ap  = kstart + p;
    int apc = min(max(ap, 0), SEQ - 1);
    float z = ((p < 80) & (ap >= 0) & (ap < SEQ)) ? 1.0f : 0.0f;
    float4 kv = *(const float4*)(kb + (size_t)apc * DM + d);
    float4 vv = *(const float4*)(vb + (size_t)apc * DM + d);
    uint2 o; o.x = pack2(kv.x * z, kv.y * z); o.y = pack2(kv.z * z, kv.w * z);
    *(uint2*)&Ks[p * 72 + d] = o;
    Vst[(d + 0) * 104 + p] = (unsigned short)bf16_rne(vv.x * z);
    Vst[(d + 1) * 104 + p] = (unsigned short)bf16_rne(vv.y * z);
    Vst[(d + 2) * 104 + p] = (unsigned short)bf16_rne(vv.z * z);
    Vst[(d + 3) * 104 + p] = (unsigned short)bf16_rne(vv.w * z);
  }
  __syncthreads();

  // ---- GEMM1: scores(16x96) = Q(16x64) @ K^T ----
  v8f sc[6] = {};
  #pragma unroll
  for (int kk = 0; kk < 64; kk += 32) {
    v16us a;
    const int arow = lane & 15;
    const int akb  = kk + ((lane < 16) ? 0 : 8);
    #pragma unroll
    for (int j = 0; j < 8; ++j) {
      a[j]     = Qs[arow * 72 + akb + j];
      a[8 + j] = Qs[arow * 72 + akb + 16 + j];
    }
    #pragma unroll
    for (int nt = 0; nt < 6; ++nt) {
      v16us bf;
      const int p  = nt * 16 + (lane & 15);       // B column = key slot
      const int dk = kk + ((lane < 16) ? 0 : 8);  // contraction = head dim
      #pragma unroll
      for (int j = 0; j < 16; ++j)
        bf[j] = Ks[p * 72 + dk + j + (j & 8)];
      sc[nt] = wmma_bf16(a, bf, sc[nt]);
    }
  }

  // ---- band mask + softmax (row M lives in one 16-lane group) ----
  const float scale = 0.125f;                 // 1/sqrt(64)
  const int   madd  = (lane < 16) ? 0 : 8;
  const int   pcol  = lane & 15;
  #pragma unroll
  for (int r = 0; r < 8; ++r) {
    const int M = r + madd;                   // query row within tile
    float x[6];
    #pragma unroll
    for (int nt = 0; nt < 6; ++nt) {
      const int p  = nt * 16 + pcol;          // relative key slot
      const int ap = kstart + p;              // absolute key position
      bool valid = (p >= M) & (p <= M + 64) & (ap >= 0) & (ap < SEQ);
      x[nt] = valid ? sc[nt][r] * scale : -1.0e30f;
    }
    float m = x[0];
    #pragma unroll
    for (int nt = 1; nt < 6; ++nt) m = fmaxf(m, x[nt]);
    #pragma unroll
    for (int off = 8; off >= 1; off >>= 1)
      m = fmaxf(m, __shfl_xor(m, off, 32));   // stays inside 16-lane group
    float ssum = 0.0f;
    #pragma unroll
    for (int nt = 0; nt < 6; ++nt) { x[nt] = __expf(x[nt] - m); ssum += x[nt]; }
    #pragma unroll
    for (int off = 8; off >= 1; off >>= 1)
      ssum += __shfl_xor(ssum, off, 32);
    const float inv = 1.0f / ssum;
    #pragma unroll
    for (int nt = 0; nt < 6; ++nt)
      Ps[M * 104 + nt * 16 + pcol] = (unsigned short)bf16_rne(x[nt] * inv);
  }
  __syncthreads();

  // ---- GEMM2: out(16x64) = P(16x96) @ V ----
  v8f oc[4] = {};
  #pragma unroll
  for (int kk = 0; kk < 96; kk += 32) {
    v16us a;
    const int arow = lane & 15;
    const int akb  = kk + ((lane < 16) ? 0 : 8);
    #pragma unroll
    for (int j = 0; j < 8; ++j) {
      a[j]     = Ps[arow * 104 + akb + j];
      a[8 + j] = Ps[arow * 104 + akb + 16 + j];
    }
    #pragma unroll
    for (int dt = 0; dt < 4; ++dt) {
      v16us bf;
      const int d  = dt * 16 + (lane & 15);       // B column = head dim
      const int pk = kk + ((lane < 16) ? 0 : 8);  // contraction = key slot
      #pragma unroll
      for (int j = 0; j < 16; ++j)
        bf[j] = Vst[d * 104 + pk + j + (j & 8)];
      oc[dt] = wmma_bf16(a, bf, oc[dt]);
    }
  }

  // ---- fused mean-pool: sum 16 rows, one atomicAdd per output dim ----
  #pragma unroll
  for (int dt = 0; dt < 4; ++dt) {
    float ps = 0.0f;
    #pragma unroll
    for (int r = 0; r < 8; ++r) ps += oc[dt][r];  // 8 rows in this lane group
    ps += __shfl_xor(ps, 16, 32);                 // + other lane group's rows
    if (lane < 16)
      atomicAdd(&pooled[b * DM + h * HD + dt * 16 + lane], ps);
  }
}

// ---------------------------------------------------------------------------
// Kernel 3: out = relu((pooled / S) @ Wfc + bfc)   (2 x 768) -- tiny
// ---------------------------------------------------------------------------
__global__ __launch_bounds__(256)
void fc_relu_kernel(const float* __restrict__ pooled,
                    const float* __restrict__ Wfc,
                    const float* __restrict__ bfc,
                    float* __restrict__ out)
{
  int idx = blockIdx.x * blockDim.x + threadIdx.x;
  if (idx >= BATCH * DM) return;
  int b = idx / DM, n = idx % DM;
  const float invS = 1.0f / (float)SEQ;
  float accum = bfc[n];
  for (int kx = 0; kx < DM; ++kx)
    accum += (pooled[b * DM + kx] * invS) * Wfc[(size_t)kx * DM + n];
  out[idx] = fmaxf(accum, 0.0f);
}

// ---------------------------------------------------------------------------
extern "C" void kernel_launch(void* const* d_in, const int* in_sizes, int n_in,
                              void* d_out, int out_size, void* d_ws, size_t ws_size,
                              hipStream_t stream)
{
  (void)in_sizes; (void)n_in; (void)out_size; (void)ws_size;

  const float* X   = (const float*)d_in[0];
  const float* Wq  = (const float*)d_in[1];
  const float* bq  = (const float*)d_in[2];
  const float* Wk  = (const float*)d_in[3];
  const float* bk  = (const float*)d_in[4];
  const float* Wv  = (const float*)d_in[5];
  const float* bv  = (const float*)d_in[6];
  const float* Wfc = (const float*)d_in[7];
  const float* bfc = (const float*)d_in[8];
  float* out = (float*)d_out;

  const size_t msd = (size_t)BATCH * SEQ * DM;   // 1,572,864 floats each
  float* qbuf   = (float*)d_ws;
  float* kbuf   = qbuf + msd;
  float* vbuf   = kbuf + msd;
  float* pooled = vbuf + msd;                    // BATCH*DM floats

  zero_kernel<<<dim3((BATCH * DM + 255) / 256), 256, 0, stream>>>(pooled, BATCH * DM);

  qkv_proj_kernel<<<dim3((BATCH * SEQ) / 64, DM / 64, 3), 256, 0, stream>>>(
      X, Wq, bq, Wk, bk, Wv, bv, qbuf, kbuf, vbuf);

  attn_pool_kernel<<<dim3(SEQ / 16, NH, BATCH), 32, 0, stream>>>(
      qbuf, kbuf, vbuf, pooled);

  fc_relu_kernel<<<dim3((BATCH * DM + 255) / 256), 256, 0, stream>>>(
      pooled, Wfc, bfc, out);
}
